// ChromaticMemory_4526895530629
// MI455X (gfx1250) — matrix operand order repaired and checked by reading, not compiled, not run on hardware
//
#include <hip/hip_runtime.h>
#include <hip/hip_bf16.h>

// ---------------------------------------------------------------------------
// ChromaticMemory retrieve_similar for MI455X (gfx1250, wave32, WMMA).
//
// combined[b,n] = strength[n] * (0.4*cos(q,mq[n]) + 0.4*cos(q,mr[n]) + 0.2*cos(q,mt[n]))
//              = qhat[b] . V[n],   V[n] = strength[n]*(0.4*mq_hat[n]+0.4*mr_hat[n]+0.2*mt_hat[n])
//
// Phase 1: qhat -> bf16            (tiny)
// Phase 2: V    -> bf16            (single 768MB fp32 HBM pass; V=128MB, L2-resident)
// Phase 3: scores = qhat.V^T via v_wmma_f32_16x16x32_bf16, masked (>0.3 ? x : -1)
// Phase 4: per-row top-5 (value desc, tie -> lower index, matching jax.lax.top_k)
// ---------------------------------------------------------------------------

typedef __bf16 bf16_t;
typedef bf16_t v16bf __attribute__((ext_vector_type(16)));
typedef bf16_t v8bf  __attribute__((ext_vector_type(8)));
typedef bf16_t v4bf  __attribute__((ext_vector_type(4)));
typedef float  v8f   __attribute__((ext_vector_type(8)));

#define D_DIM 512
#define EPS_F 1e-8f
#define SIM_THRESHOLD 0.3f
#define TOPK 5

// Load one 512-float row (16 floats/lane as 4x float4, coalesced) and return the
// wave-wide sum of squares (cross-lane reduction over the 32 lanes).
__device__ __forceinline__ float load_row_sumsq(const float* __restrict__ p, int lane, float4 x[4]) {
    float ss = 0.f;
#pragma unroll
    for (int c = 0; c < 4; c++) {
        x[c] = *(const float4*)(p + c * 128 + lane * 4);
        ss += x[c].x * x[c].x + x[c].y * x[c].y + x[c].z * x[c].z + x[c].w * x[c].w;
    }
#pragma unroll
    for (int off = 16; off > 0; off >>= 1)
        ss += __shfl_xor(ss, off, 32);
    return ss;
}

__device__ __forceinline__ void scale_acc(float4 acc[4], const float4 x[4], float sc) {
#pragma unroll
    for (int c = 0; c < 4; c++) {
        acc[c].x += x[c].x * sc;
        acc[c].y += x[c].y * sc;
        acc[c].z += x[c].z * sc;
        acc[c].w += x[c].w * sc;
    }
}

// ---------------- Phase 1: qhat = q / (||q|| + eps), stored bf16 ----------------
__global__ void prep_query_kernel(const float* __restrict__ q, bf16_t* __restrict__ Qn, int B) {
    const int lane = threadIdx.x & 31;
    const int row  = blockIdx.x * 8 + (threadIdx.x >> 5);   // one wave per row
    if (row >= B) return;
    const size_t off = (size_t)row * D_DIM;
    float4 x[4];
    float ss = load_row_sumsq(q + off, lane, x);
    float sc = 1.0f / (sqrtf(ss) + EPS_F);
#pragma unroll
    for (int c = 0; c < 4; c++) {
        v4bf o = { (bf16_t)(x[c].x * sc), (bf16_t)(x[c].y * sc),
                   (bf16_t)(x[c].z * sc), (bf16_t)(x[c].w * sc) };
        *(v4bf*)(Qn + off + c * 128 + lane * 4) = o;
    }
}

// ---------------- Phase 2: fused normalized-weighted memory matrix ----------------
__global__ void prep_memory_kernel(const float* __restrict__ mq, const float* __restrict__ mr,
                                   const float* __restrict__ mt, const float* __restrict__ strength,
                                   bf16_t* __restrict__ V, int N) {
    const int lane = threadIdx.x & 31;
    const int row  = blockIdx.x * 8 + (threadIdx.x >> 5);   // one wave per row
    if (row >= N) return;
    const size_t off = (size_t)row * D_DIM;
    float4 x[4];
    float4 acc[4] = { {0,0,0,0}, {0,0,0,0}, {0,0,0,0}, {0,0,0,0} };

    float ss = load_row_sumsq(mq + off, lane, x);
    scale_acc(acc, x, 0.4f / (sqrtf(ss) + EPS_F));
    ss = load_row_sumsq(mr + off, lane, x);
    scale_acc(acc, x, 0.4f / (sqrtf(ss) + EPS_F));
    ss = load_row_sumsq(mt + off, lane, x);
    scale_acc(acc, x, 0.2f / (sqrtf(ss) + EPS_F));

    const float st = strength[row];
#pragma unroll
    for (int c = 0; c < 4; c++) {
        v4bf o = { (bf16_t)(acc[c].x * st), (bf16_t)(acc[c].y * st),
                   (bf16_t)(acc[c].z * st), (bf16_t)(acc[c].w * st) };
        *(v4bf*)(V + off + c * 128 + lane * 4) = o;
    }
}

// ---------------- Phase 3: scores = Qn . V^T (bf16 WMMA), masked ----------------
// Block = 256 threads = 8 waves. Block tile: 16 queries x 256 candidates.
// Wave tile: 16 x 32 (two 16x16 WMMA accumulators). K loop: 512 in steps of 32.
//
// ISA VGPR layouts (CDNA5 ISA):
//  A (16x32 bf16): lanes 0-15 -> row M=lane, K {0..7,16..23}; lanes 16-31 -> K {8..15,24..31}
//  B (32x16 bf16): lanes 0-15 -> col N=lane; lanes 16-31 -> col N=lane-16; K split as A (code: A and B both use K 16h..16h+15)
//  C (16x16 f32):  lanes 0-15 -> N=lane, VGPR j = M=j; lanes 16-31 -> M=j+8
__global__ void gemm_mask_kernel(const bf16_t* __restrict__ Qn, const bf16_t* __restrict__ V,
                                 float* __restrict__ scores, int N) {
    const int lane = threadIdx.x & 31;
    const int wave = threadIdx.x >> 5;
    const int half = lane >> 4;        // 0: lanes 0-15, 1: lanes 16-31
    const int l16  = lane & 15;
    const int m_base = blockIdx.y * 16;
    const int n0     = blockIdx.x * 256 + wave * 32;

    v8f c0 = {};
    v8f c1 = {};

    const bf16_t* __restrict__ qrow  = Qn + (size_t)(m_base + l16) * D_DIM;
    const bf16_t* __restrict__ vrow0 = V + (size_t)(n0 + l16) * D_DIM;
    const bf16_t* __restrict__ vrow1 = vrow0 + (size_t)16 * D_DIM;

#pragma unroll
    for (int kk = 0; kk < D_DIM; kk += 32) {
        // A fragment: one aligned 32B chunk per lane (contiguous K-range, matching B).
        v16bf a = *(const v16bf*)(qrow + kk + half * 16);
        // B fragments: one aligned 32B chunk per lane (contiguous K-range of row n).
        v16bf b0 = *(const v16bf*)(vrow0 + kk + half * 16);
        v16bf b1 = *(const v16bf*)(vrow1 + kk + half * 16);

        c0 = __builtin_amdgcn_wmma_f32_16x16x32_bf16(false, a, false, b0, (short)0, c0, false, false);
        c1 = __builtin_amdgcn_wmma_f32_16x16x32_bf16(false, a, false, b1, (short)0, c1, false, false);
    }

    const int col0 = n0 + l16;
#pragma unroll
    for (int j = 0; j < 8; j++) {
        const int m = m_base + j + half * 8;
        const float s0 = c0[j];
        const float s1 = c1[j];
        scores[(size_t)m * N + col0]      = (s0 > SIM_THRESHOLD) ? s0 : -1.0f;
        scores[(size_t)m * N + col0 + 16] = (s1 > SIM_THRESHOLD) ? s1 : -1.0f;
    }
}

// ---------------- Phase 4: per-row top-5 (desc value, tie -> lower index) ----------------
__global__ void topk5_kernel(const float* __restrict__ scores, float* __restrict__ outv,
                             int* __restrict__ outi, int N) {
    __shared__ float sv[256 * TOPK];
    __shared__ int   si[256 * TOPK];
    const int t = threadIdx.x;
    const int b = blockIdx.x;

    float v[TOPK];
    int   id[TOPK];
#pragma unroll
    for (int j = 0; j < TOPK; j++) { v[j] = -3.0e38f; id[j] = 0x7fffffff; }

    const float* __restrict__ row = scores + (size_t)b * N;
    // Ascending index order per thread + strict '>' insertion keeps earliest index on ties.
    for (int i = t; i < N; i += 256) {
        const float nv = row[i];
        if (nv > v[TOPK - 1]) {
            v[TOPK - 1] = nv;
            id[TOPK - 1] = i;
#pragma unroll
            for (int p = TOPK - 1; p > 0; p--) {
                if (v[p] > v[p - 1]) {
                    float tv = v[p]; v[p] = v[p - 1]; v[p - 1] = tv;
                    int ti = id[p]; id[p] = id[p - 1]; id[p - 1] = ti;
                }
            }
        }
    }
#pragma unroll
    for (int j = 0; j < TOPK; j++) { sv[t * TOPK + j] = v[j]; si[t * TOPK + j] = id[j]; }

    for (int s = 128; s > 0; s >>= 1) {
        __syncthreads();
        if (t < s) {
            float av[TOPK], bv[TOPK], mv[TOPK];
            int   ai[TOPK], bi[TOPK], mi[TOPK];
#pragma unroll
            for (int j = 0; j < TOPK; j++) {
                av[j] = sv[t * TOPK + j];       ai[j] = si[t * TOPK + j];
                bv[j] = sv[(t + s) * TOPK + j]; bi[j] = si[(t + s) * TOPK + j];
            }
            int ia = 0, ib = 0;
#pragma unroll
            for (int j = 0; j < TOPK; j++) {
                const bool take_a = (av[ia] > bv[ib]) ||
                                    (av[ia] == bv[ib] && ai[ia] <= bi[ib]);
                if (take_a) { mv[j] = av[ia]; mi[j] = ai[ia]; ia++; }
                else        { mv[j] = bv[ib]; mi[j] = bi[ib]; ib++; }
            }
#pragma unroll
            for (int j = 0; j < TOPK; j++) { sv[t * TOPK + j] = mv[j]; si[t * TOPK + j] = mi[j]; }
        }
    }
    __syncthreads();
    if (t == 0) {
#pragma unroll
        for (int j = 0; j < TOPK; j++) {
            outv[b * TOPK + j] = sv[j];
            outi[b * TOPK + j] = si[j];
        }
    }
}

extern "C" void kernel_launch(void* const* d_in, const int* in_sizes, int n_in,
                              void* d_out, int out_size, void* d_ws, size_t ws_size,
                              hipStream_t stream) {
    (void)n_in; (void)out_size; (void)ws_size;
    const float* query = (const float*)d_in[0];
    const float* mq    = (const float*)d_in[1];
    const float* mr    = (const float*)d_in[2];
    const float* mt    = (const float*)d_in[3];
    const float* ms    = (const float*)d_in[4];
    // d_in[5] = top_k (device scalar); reference uses 5, TOPK is compile-time.

    const int B = in_sizes[0] / D_DIM;   // 256
    const int N = in_sizes[4];           // 131072

    // Workspace layout: V bf16 [N*D] | scores f32 [B*N] | Qn bf16 [B*D]
    char* ws = (char*)d_ws;
    bf16_t* V      = (bf16_t*)ws;
    float*  scores = (float*)(ws + (size_t)N * D_DIM * sizeof(bf16_t));
    bf16_t* Qn     = (bf16_t*)(ws + (size_t)N * D_DIM * sizeof(bf16_t)
                                  + (size_t)B * N * sizeof(float));

    float* outv = (float*)d_out;                    // [B, 5] values
    int*   outi = (int*)(outv + (size_t)B * TOPK);  // [B, 5] indices (int32)

    prep_query_kernel<<<(B + 7) / 8, 256, 0, stream>>>(query, Qn, B);
    prep_memory_kernel<<<(N + 7) / 8, 256, 0, stream>>>(mq, mr, mt, ms, V, N);

    dim3 gg(N / 256, B / 16);
    gemm_mask_kernel<<<gg, 256, 0, stream>>>(Qn, V, scores, N);

    topk5_kernel<<<B, 256, 0, stream>>>(scores, outv, outi, N);
}
